// RecurrentConservationOfEnergyModel_996432413562
// MI455X (gfx1250) — matrix operand order repaired and checked by reading, not compiled, not run on hardware
//
#include <hip/hip_runtime.h>

// ---------------------------------------------------------------------------
// Model constants
// ---------------------------------------------------------------------------
#define NBE   256     // B*E rows
#define CCH   128     // channels / latent
#define NFR   512     // frames
#define NIF   128     // impulse frames
#define STEPC 64
#define IMPN  8192
#define EPSU  1e-8f

typedef __attribute__((ext_vector_type(16))) _Float16 v16h;
typedef __attribute__((ext_vector_type(8)))  float    v8f;

#define DEV __device__ __forceinline__

// ---------------------------------------------------------------------------
// CDNA5 async global->LDS (ASYNCcnt) helpers.
//   global_load_async_to_lds_b128: per-lane copy of 16 bytes from global vaddr to LDS offset vdst,
//   fire-and-forget (no VGPR data), drained with s_wait_asynccnt.
// ---------------------------------------------------------------------------
typedef __attribute__((address_space(3))) char lds_char_t;

DEV unsigned lds_byte_offset(const void* p) {
  return (unsigned)(unsigned long long)(lds_char_t*)p;   // generic -> AS(3) -> offset
}

DEV void async_g2l_b128(unsigned lds_off, const void* g) {
  asm volatile("global_load_async_to_lds_b128 %0, %1, off"
               :: "v"(lds_off), "v"(g) : "memory");
}

DEV void wait_asynccnt0() {
  asm volatile("s_wait_asynccnt 0x0" ::: "memory");
}

// ---------------------------------------------------------------------------
// WMMA fragment builders (wave32, 16x16x32 f16 -> f32)
//   A: 16(M) x 32(K) f16 tile from row-major [16][ld] array
//   B: 32(K) x 16(N) tile; source W is row-major (out,in) so B[k][n]=W[n][k]
//   C/D: VGPR r of lane l holds (M = r + 8*(l>>4), N = l&15)
// ---------------------------------------------------------------------------
DEV v16h load_a(const _Float16* X, int ld, int k0, int lane) {
  const int m  = lane & 15;
  const int hi = (lane >> 4) ? 16 : 0;
  const _Float16* row = X + (size_t)m * ld + k0;
  v16h a;
#pragma unroll
  for (int v = 0; v < 8; ++v) {
    int k = hi + 2 * v;
    a[2 * v]     = row[k];
    a[2 * v + 1] = row[k + 1];
  }
  return a;
}

DEV v16h load_b(const _Float16* W, int ld, int n0, int k0, int lane) {
  const int n  = lane & 15;
  const int hi = (lane >> 4) ? 16 : 0;
  const _Float16* row = W + (size_t)(n0 + n) * ld + k0 + hi;
  v16h b;
#pragma unroll
  for (int v = 0; v < 8; ++v) { b[2 * v] = row[2 * v]; b[2 * v + 1] = row[2 * v + 1]; }
  return b;
}

// B fragment where W element (n,i) sits at W[n*128*KW + i*KW + ktap]  (conv taps)
DEV v16h load_b_strided(const _Float16* W, int KW, int ktap, int ibase, int n0, int lane) {
  const int n  = n0 + (lane & 15);
  const int hi = (lane >> 4) ? 16 : 0;
  const _Float16* base = W + (size_t)n * 128 * KW + (size_t)(ibase + hi) * KW + ktap;
  v16h b;
#pragma unroll
  for (int v = 0; v < 8; ++v) { b[2 * v] = base[(2 * v) * KW]; b[2 * v + 1] = base[(2 * v + 1) * KW]; }
  return b;
}

DEV v8f wmma_f16(v16h a, v16h b, v8f c) {
  return __builtin_amdgcn_wmma_f32_16x16x32_f16(false, a, false, b, (short)0, c, false, false);
}

// Row reduction: 16 threads per row (cs = 0..15), everyone gets the sum.
DEV float rowsum16(float* red, int r, int cs, float part) {
  __syncthreads();
  red[r * 16 + cs] = part;
  __syncthreads();
  float s = 0.f;
#pragma unroll
  for (int i = 0; i < 16; ++i) s += red[r * 16 + i];
  return s;
}

// ---------------------------------------------------------------------------
// 4-layer MLP stack on a 16x128 tile resident in LDS.
//   xh   : LDS f16 [16][128]  input (clobbered layer by layer)
//   ybuf : LDS f32 [16][128]  final linear output (no activation)
//   layers 0..2: linear + leaky(0.2) + layernorm ; layer 3: linear only
// Caller must have synced after last xh write.
// ---------------------------------------------------------------------------
DEV void stack_tile(const _Float16* W0, const _Float16* W1, const _Float16* W2,
                    const _Float16* W3,
                    const float* bs0, const float* bs1, const float* bs2,
                    const float* ls0, const float* ls1, const float* ls2,
                    const float* lb0, const float* lb1, const float* lb2,
                    const float* ob,
                    _Float16* xh, float* ybuf, float* red, int tid) {
  const int lane = tid & 31, n0 = (tid >> 5) * 16;
  const int r = tid >> 4, cs = tid & 15, c0 = cs * 8;
  const int nn = n0 + (lane & 15);
  const int mrow = (lane >> 4) ? 8 : 0;
  const _Float16* Ws[3] = {W0, W1, W2};
  const float* bss[3] = {bs0, bs1, bs2};
  const float* lss[3] = {ls0, ls1, ls2};
  const float* lbs[3] = {lb0, lb1, lb2};

#pragma unroll
  for (int l = 0; l < 3; ++l) {
    v8f c = {};
#pragma unroll
    for (int k0 = 0; k0 < 128; k0 += 32)
      c = wmma_f16(load_a(xh, 128, k0, lane), load_b(Ws[l], 128, n0, k0, lane), c);
    __syncthreads();
    float bv = bss[l][nn];
#pragma unroll
    for (int rr = 0; rr < 8; ++rr) ybuf[(rr + mrow) * 128 + nn] = c[rr] + bv;
    __syncthreads();
    float v[8], s = 0.f, q = 0.f;
#pragma unroll
    for (int j = 0; j < 8; ++j) {
      float y = ybuf[r * 128 + c0 + j];
      y = (y > 0.f) ? y : 0.2f * y;              // leaky_relu(0.2)
      v[j] = y; s += y; q += y * y;
    }
    float m   = rowsum16(red, r, cs, s) * (1.f / 128.f);
    float var = rowsum16(red, r, cs, q) * (1.f / 128.f) - m * m;
    float rs  = rsqrtf(var + 1e-5f);
#pragma unroll
    for (int j = 0; j < 8; ++j)
      xh[r * 128 + c0 + j] = (_Float16)(((v[j] - m) * rs) * lss[l][c0 + j] + lbs[l][c0 + j]);
    __syncthreads();
  }
  v8f c = {};
#pragma unroll
  for (int k0 = 0; k0 < 128; k0 += 32)
    c = wmma_f16(load_a(xh, 128, k0, lane), load_b(W3, 128, n0, k0, lane), c);
  __syncthreads();
  float obv = ob[nn];
#pragma unroll
  for (int rr = 0; rr < 8; ++rr) ybuf[(rr + mrow) * 128 + nn] = c[rr] + obv;
  __syncthreads();
}

// ---------------------------------------------------------------------------
// f32 -> f16 weight conversion
// ---------------------------------------------------------------------------
__global__ void k_cvt(const float* __restrict__ s, _Float16* __restrict__ d, int n) {
  int i = blockIdx.x * 256 + threadIdx.x;
  if (i < n) d[i] = (_Float16)s[i];
}

// ---------------------------------------------------------------------------
// Generic GEMM: Y(R,Nout) = X(R,128) @ W(Nout,128)^T + bias.  16-row tiles,
// 8 waves x one 16x16 output tile each, grid.y covers Nout/128.
// ---------------------------------------------------------------------------
__global__ __launch_bounds__(256) void k_gemm_bias(const float* __restrict__ X,
                                                   const _Float16* __restrict__ W,
                                                   const float* __restrict__ bias,
                                                   float* __restrict__ Y, int Nout) {
  __shared__ _Float16 Af[16 * 128];
  const int tid = threadIdx.x;
  const int rbase = blockIdx.x * 16;
  const int r = tid >> 4, c0 = (tid & 15) * 8;
#pragma unroll
  for (int j = 0; j < 8; ++j)
    Af[r * 128 + c0 + j] = (_Float16)X[(size_t)(rbase + r) * 128 + c0 + j];
  __syncthreads();
  const int lane = tid & 31;
  const int n0 = blockIdx.y * 128 + (tid >> 5) * 16;
  v8f c = {};
#pragma unroll
  for (int k0 = 0; k0 < 128; k0 += 32)
    c = wmma_f16(load_a(Af, 128, k0, lane), load_b(W, 128, n0, k0, lane), c);
  const int nn = n0 + (lane & 15);
  const int mrow = (lane >> 4) ? 8 : 0;
  float bv = bias[nn];
#pragma unroll
  for (int rr = 0; rr < 8; ++rr)
    Y[(size_t)(rbase + rr + mrow) * Nout + nn] = c[rr] + bv;
}

// ---------------------------------------------------------------------------
// Upsampling conv (lhs_dilation=2, pad (2,2), kernel 4) as implicit GEMM.
//   out even t=2j : W[:,:,0]@x[j-1] + W[:,:,2]@x[j]
//   out odd  t=2j+1: W[:,:,1]@x[j]   + W[:,:,3]@x[j+1]
// K = 2 taps * 128 channels = 256. Rows tiled over (n,j) per parity.
// ---------------------------------------------------------------------------
__global__ __launch_bounds__(256) void k_conv_dil(const float* __restrict__ xin,
                                                  const _Float16* __restrict__ Wh,
                                                  const float* __restrict__ cb,
                                                  float* __restrict__ xout, int Lin) {
  __shared__ _Float16 Af[16 * 256];
  const int tid = threadIdx.x;
  const int p = blockIdx.y;
  const int rbase = blockIdx.x * 16;
  const int r = tid >> 4, kseg = (tid & 15) * 16;
  {
    int row = rbase + r, n = row / Lin, j = row % Lin;
#pragma unroll
    for (int u = 0; u < 16; ++u) {
      int kk = kseg + u, tap = kk >> 7, i = kk & 127;
      int jj = (p == 0) ? (tap ? j : j - 1) : (tap ? j + 1 : j);
      float v = (jj >= 0 && jj < Lin) ? xin[((size_t)n * 128 + i) * Lin + jj] : 0.f;
      Af[r * 256 + kk] = (_Float16)v;
    }
  }
  __syncthreads();
  const int lane = tid & 31, n0 = (tid >> 5) * 16;
  v8f c = {};
#pragma unroll
  for (int k0 = 0; k0 < 256; k0 += 32) {
    int tap = k0 >> 7;
    int ktap = (p == 0) ? (tap ? 2 : 0) : (tap ? 3 : 1);
    c = wmma_f16(load_a(Af, 256, k0, lane),
                 load_b_strided(Wh, 4, ktap, k0 & 127, n0, lane), c);
  }
  const int Lout = Lin * 2;
  const int nn = n0 + (lane & 15);
  const int mrow = (lane >> 4) ? 8 : 0;
  float bias = cb[nn];
#pragma unroll
  for (int rr = 0; rr < 8; ++rr) {
    int row = rbase + rr + mrow, n = row / Lin, j = row % Lin, t = 2 * j + p;
    xout[((size_t)n * 128 + nn) * Lout + t] = c[rr] + bias;
  }
}

// Final conv: kernel 3, pad 1. K = 3 taps * 128 = 384. Rows over (n,t).
__global__ __launch_bounds__(256) void k_conv_final(const float* __restrict__ xin,
                                                    const _Float16* __restrict__ Wh,
                                                    const float* __restrict__ ob,
                                                    float* __restrict__ dout) {
  __shared__ _Float16 Af[16 * 384];
  const int tid = threadIdx.x;
  const int rbase = blockIdx.x * 16;
  const int r = tid >> 4, kseg = (tid & 15) * 24;
  {
    int row = rbase + r, n = row >> 9, t = row & 511;
#pragma unroll
    for (int u = 0; u < 24; ++u) {
      int kk = kseg + u, tap = kk >> 7, i = kk & 127;
      int jj = t - 1 + tap;
      float v = (jj >= 0 && jj < 512) ? xin[((size_t)n * 128 + i) * 512 + jj] : 0.f;
      Af[r * 384 + kk] = (_Float16)v;
    }
  }
  __syncthreads();
  const int lane = tid & 31, n0 = (tid >> 5) * 16;
  v8f c = {};
#pragma unroll
  for (int k0 = 0; k0 < 384; k0 += 32)
    c = wmma_f16(load_a(Af, 384, k0, lane),
                 load_b_strided(Wh, 3, k0 >> 7, k0 & 127, n0, lane), c);
  const int nn = n0 + (lane & 15);
  const int mrow = (lane >> 4) ? 8 : 0;
  float bias = ob[nn];
#pragma unroll
  for (int rr = 0; rr < 8; ++rr) {
    int row = rbase + rr + mrow, n = row >> 9, t = row & 511;
    dout[((size_t)n * 128 + nn) * 512 + t] = c[rr] + bias;
  }
}

// Batch-norm stats over (N, L) per channel (mean + inv-std).
__global__ void k_bn_stats(const float* __restrict__ x, int L,
                           float* __restrict__ mean, float* __restrict__ rstd) {
  const int c = blockIdx.x, tid = threadIdx.x;
  const int total = 256 * L;
  float s = 0.f, q = 0.f;
  for (int e = tid; e < total; e += 256) {
    int n = e / L, t = e % L;
    float v = x[((size_t)n * 128 + c) * L + t];
    s += v; q += v * v;
  }
  __shared__ float rs[256], rq[256];
  rs[tid] = s; rq[tid] = q; __syncthreads();
  for (int o = 128; o > 0; o >>= 1) {
    if (tid < o) { rs[tid] += rs[tid + o]; rq[tid] += rq[tid + o]; }
    __syncthreads();
  }
  if (tid == 0) {
    float m = rs[0] / total, var = rq[0] / total - m * m;
    mean[c] = m; rstd[c] = rsqrtf(var + 1e-5f);
  }
}

// BN apply + leaky(0.2), in place.  Layout (n,c,t) flat.
__global__ void k_bn_apply(float* __restrict__ x, int L,
                           const float* __restrict__ mean, const float* __restrict__ rstd,
                           const float* __restrict__ s, const float* __restrict__ b) {
  size_t idx = (size_t)blockIdx.x * 256 + threadIdx.x;
  int c = (int)((idx / L) & 127);
  float v = x[idx];
  v = (v - mean[c]) * rstd[c] * s[c] + b[c];
  x[idx] = (v > 0.f) ? v : 0.2f * v;
}

// unit_norm over channel axis of d (n,c,t), transpose to dt(n,t,c), fused leak.
__global__ __launch_bounds__(128) void k_unorm_leak(const float* __restrict__ d,
                                                    const float* __restrict__ lw,
                                                    const float* __restrict__ lb,
                                                    float* __restrict__ dt,
                                                    float* __restrict__ leak) {
  const int row = blockIdx.x;               // n*512 + t
  const int n = row >> 9, t = row & 511, c = threadIdx.x;
  float v = d[((size_t)n * 128 + c) * 512 + t];
  __shared__ float red[128];
  red[c] = v * v; __syncthreads();
  for (int o = 64; o > 0; o >>= 1) { if (c < o) red[c] += red[c + o]; __syncthreads(); }
  float nv = v / (sqrtf(red[0]) + EPSU);
  dt[(size_t)row * 128 + c] = nv;
  __syncthreads();
  red[c] = nv * lw[c]; __syncthreads();
  for (int o = 64; o > 0; o >>= 1) { if (c < o) red[c] += red[c + o]; __syncthreads(); }
  if (c == 0) leak[row] = 0.1f + 0.98f / (1.f + expf(-(red[0] + lb[0])));
}

// ---------------------------------------------------------------------------
// Impulse frames -> embed stack -> emb (n, jf, c)
// ---------------------------------------------------------------------------
__global__ __launch_bounds__(256) void k_frames(const float* __restrict__ imp,
    const _Float16* W0, const _Float16* W1, const _Float16* W2, const _Float16* W3,
    const float* bs0, const float* bs1, const float* bs2,
    const float* ls0, const float* ls1, const float* ls2,
    const float* lb0, const float* lb1, const float* lb2,
    const float* ob, float* __restrict__ embout) {
  __shared__ _Float16 xh[16 * 128];
  __shared__ float ybuf[16 * 128];
  __shared__ float red[256];
  const int tid = threadIdx.x;
  const int rbase = blockIdx.x * 16;
  const int r = tid >> 4, cs = tid & 15, c0 = cs * 8;
  const int row = rbase + r, n = row >> 7, jf = row & 127;
  float q = 0.f;
#pragma unroll
  for (int j = 0; j < 8; ++j) {
    int c = c0 + j, idx = jf * STEPC + c;
    float v = (idx < IMPN) ? imp[(size_t)n * IMPN + idx] : 0.f;
    float ham = 0.54f - 0.46f * cosf(6.28318530718f * c * (1.f / 128.f));
    v *= ham;
    xh[r * 128 + c] = (_Float16)v;
    q += v * v;
  }
  float frn = sqrtf(rowsum16(red, r, cs, q));   // includes the needed barriers
  stack_tile(W0, W1, W2, W3, bs0, bs1, bs2, ls0, ls1, ls2, lb0, lb1, lb2, ob,
             xh, ybuf, red, tid);
  float v[8], q2 = 0.f;
#pragma unroll
  for (int j = 0; j < 8; ++j) { v[j] = ybuf[r * 128 + c0 + j]; q2 += v[j] * v[j]; }
  float sc = frn / (sqrtf(rowsum16(red, r, cs, q2)) + EPSU);
#pragma unroll
  for (int j = 0; j < 8; ++j) embout[(size_t)row * 128 + c0 + j] = v[j] * sc;
}

// ---------------------------------------------------------------------------
// Recurrent scan: 16 blocks x 16 rows, 512 steps, both stacks' weights
// LDS-resident in f16 (256 KB of CDNA5's 320KB/WGP), staged via
// GLOBAL_LOAD_ASYNC_TO_LDS (ASYNCcnt), overlap-add fused.
// ---------------------------------------------------------------------------
#define SCAN_SMEM_BYTES (8*128*128*2 + (16*128 + 16*128 + 256 + 16*64)*4 + 16*128*2)

__global__ __launch_bounds__(256) void k_scan(
    const float* __restrict__ emb, const float* __restrict__ wpre,
    const float* __restrict__ bpre, const float* __restrict__ leak,
    const _Float16* gW,                              // 8 contiguous 128x128 f16 mats
    const float* obs0, const float* obs1, const float* obs2,
    const float* ols0, const float* ols1, const float* ols2,
    const float* olb0, const float* olb1, const float* olb2, const float* oob,
    const float* ubs0, const float* ubs1, const float* ubs2,
    const float* uls0, const float* uls1, const float* uls2,
    const float* ulb0, const float* ulb1, const float* ulb2, const float* uob,
    float* __restrict__ out) {
  extern __shared__ char smem[];
  _Float16* Wall  = (_Float16*)smem;                 // 8 matrices, 131072 halves
  float*    hbuf  = (float*)(smem + 8 * 128 * 128 * 2);
  float*    ybuf  = hbuf + 16 * 128;
  float*    red   = ybuf + 16 * 128;
  float*    prevb = red + 256;
  _Float16* xh    = (_Float16*)(prevb + 16 * 64);

  const int tid = threadIdx.x;
  const int r = tid >> 4, cs = tid & 15, c0 = cs * 8;
  const int n = blockIdx.x * 16 + r;                 // this thread's global row

  // Stage both stacks' weights (256 KB) into LDS with async DMA: each lane
  // issues 16-byte async copies; ASYNCcnt drained once, then barrier.
  {
    const unsigned wall0 = lds_byte_offset(Wall);
    const char* src = (const char*)gW;
    for (unsigned i = (unsigned)tid * 16u; i < 8u * 128u * 128u * 2u; i += 256u * 16u)
      async_g2l_b128(wall0 + i, src + i);
  }

#pragma unroll
  for (int j = 0; j < 8; ++j) hbuf[r * 128 + c0 + j] = 0.f;
  for (int i = tid; i < 16 * 64; i += 256) prevb[i] = 0.f;

  wait_asynccnt0();          // async weight DMA complete (overlapped with init)
  __syncthreads();

  const _Float16* Wol[4]  = {Wall, Wall + 16384, Wall + 32768, Wall + 49152};
  const _Float16* Wout[4] = {Wall + 65536, Wall + 81920, Wall + 98304, Wall + 114688};

  float hamw[8];
#pragma unroll
  for (int j = 0; j < 8; ++j)
    hamw[j] = 0.54f - 0.46f * cosf(6.28318530718f * (c0 + j) * (1.f / 128.f));

  for (int t = 0; t < 512; ++t) {
    const float* wrow = &wpre[((size_t)n * 512 + t) * 128 + c0];
    const float* brow = &bpre[((size_t)n * 512 + t) * 128 + c0];
    if (t < 511) {                                   // -> global_prefetch_b8
      __builtin_prefetch(&wpre[((size_t)n * 512 + t + 1) * 128 + c0], 0, 1);
      __builtin_prefetch(&bpre[((size_t)n * 512 + t + 1) * 128 + c0], 0, 1);
    }
    // h = h + e
    float h[8], q = 0.f;
#pragma unroll
    for (int j = 0; j < 8; ++j) {
      float e = (t < NIF) ? emb[((size_t)n * NIF + t) * 128 + c0 + j] : 0.f;
      h[j] = hbuf[r * 128 + c0 + j] + e;
      q += h[j] * h[j];
    }
    float cn = sqrtf(rowsum16(red, r, cs, q));
    // h = unit_norm(h*w + b) * cn
    float tmp[8]; q = 0.f;
#pragma unroll
    for (int j = 0; j < 8; ++j) { tmp[j] = h[j] * wrow[j] + brow[j]; q += tmp[j] * tmp[j]; }
    float sc = cn / (sqrtf(rowsum16(red, r, cs, q)) + EPSU);
#pragma unroll
    for (int j = 0; j < 8; ++j) {
      h[j] = tmp[j] * sc;
      xh[r * 128 + c0 + j] = (_Float16)h[j];
    }
    __syncthreads();
    // ol_pre = stack_ol(h)
    stack_tile(Wol[0], Wol[1], Wol[2], Wol[3], obs0, obs1, obs2,
               ols0, ols1, ols2, olb0, olb1, olb2, oob, xh, ybuf, red, tid);
    float lk = leak[(size_t)n * 512 + t];
    float y[8]; q = 0.f;
#pragma unroll
    for (int j = 0; j < 8; ++j) { y[j] = ybuf[r * 128 + c0 + j]; q += y[j] * y[j]; }
    float osc = (cn * lk) / (sqrtf(rowsum16(red, r, cs, q)) + EPSU);
    float ol[8]; q = 0.f;
#pragma unroll
    for (int j = 0; j < 8; ++j) { ol[j] = y[j] * osc; q += ol[j] * ol[j]; }
    float on = sqrtf(rowsum16(red, r, cs, q));
    // h = unit_norm(h - ol) * (cn - on)
    float d8[8]; q = 0.f;
#pragma unroll
    for (int j = 0; j < 8; ++j) { d8[j] = h[j] - ol[j]; q += d8[j] * d8[j]; }
    float hs = (cn - on) / (sqrtf(rowsum16(red, r, cs, q)) + EPSU);
#pragma unroll
    for (int j = 0; j < 8; ++j) {
      hbuf[r * 128 + c0 + j] = d8[j] * hs;           // next-step state (thread-private slots)
      xh[r * 128 + c0 + j]   = (_Float16)ol[j];
    }
    __syncthreads();
    // of = unit_norm(stack_out(ol) * ham) * on
    stack_tile(Wout[0], Wout[1], Wout[2], Wout[3], ubs0, ubs1, ubs2,
               uls0, uls1, uls2, ulb0, ulb1, ulb2, uob, xh, ybuf, red, tid);
    float g[8]; q = 0.f;
#pragma unroll
    for (int j = 0; j < 8; ++j) { g[j] = ybuf[r * 128 + c0 + j] * hamw[j]; q += g[j] * g[j]; }
    float ofs = on / (sqrtf(rowsum16(red, r, cs, q)) + EPSU);
    // overlap-add: out[t*64+k] = of[k] + prev_b[k];  prev_b = of[64:]
    float pb[8];
    if (c0 < STEPC) {
#pragma unroll
      for (int j = 0; j < 8; ++j) pb[j] = prevb[r * STEPC + c0 + j];
    }
    __syncthreads();
    if (c0 < STEPC) {
#pragma unroll
      for (int j = 0; j < 8; ++j)
        out[(size_t)n * (NFR * STEPC) + (size_t)t * STEPC + c0 + j] = g[j] * ofs + pb[j];
    } else {
#pragma unroll
      for (int j = 0; j < 8; ++j) prevb[r * STEPC + (c0 - STEPC) + j] = g[j] * ofs;
    }
    __syncthreads();
  }
}

// ---------------------------------------------------------------------------
// Host orchestration
// ---------------------------------------------------------------------------
extern "C" void kernel_launch(void* const* d_in, const int* in_sizes, int n_in,
                              void* d_out, int out_size, void* d_ws, size_t ws_size,
                              hipStream_t stream) {
  (void)in_sizes; (void)n_in; (void)out_size; (void)ws_size;
  auto f = [&](int i) { return (const float*)d_in[i]; };

  // jax-pytree (sorted-key) flatten order of setup_inputs()
  const float* embedding = f(0);
  const float* impulse   = f(1);
  const float* p_b_b = f(2);
  const float* p_b_w = f(3);
  const float* e_bs[3]  = {f(4), f(5), f(6)};
  const float* e_lnb[3] = {f(7), f(8), f(9)};
  const float* e_lns[3] = {f(10), f(11), f(12)};
  const float* e_outb = f(13); const float* e_outw = f(14);
  const float* e_ws[3] = {f(15), f(16), f(17)};
  const float* p_l_b = f(18);
  const float* p_l_w = f(19);
  const float* o_bs[3]  = {f(20), f(21), f(22)};
  const float* o_lnb[3] = {f(23), f(24), f(25)};
  const float* o_lns[3] = {f(26), f(27), f(28)};
  const float* o_outb = f(29); const float* o_outw = f(30);
  const float* o_ws[3] = {f(31), f(32), f(33)};
  const float* u_bs[3]  = {f(34), f(35), f(36)};
  const float* u_lnb[3] = {f(37), f(38), f(39)};
  const float* u_lns[3] = {f(40), f(41), f(42)};
  const float* u_outb = f(43); const float* u_outw = f(44);
  const float* u_ws[3] = {f(45), f(46), f(47)};
  const float* up_bnb[6] = {f(48), f(49), f(50), f(51), f(52), f(53)};
  const float* up_bns[6] = {f(54), f(55), f(56), f(57), f(58), f(59)};
  const float* up_cb[6]  = {f(60), f(61), f(62), f(63), f(64), f(65)};
  const float* up_cw[6]  = {f(66), f(67), f(68), f(69), f(70), f(71)};
  const float* up_linb = f(72); const float* up_linw = f(73);
  const float* up_ob = f(74);   const float* up_ow = f(75);
  const float* p_w_b = f(76);   const float* p_w_w = f(77);

  // workspace partition
  char* ws = (char*)d_ws; size_t off = 0;
  auto alloc = [&](size_t bytes) { void* p = ws + off; off = (off + bytes + 255) & ~(size_t)255; return p; };
  const size_t ACTN = (size_t)NBE * 128 * 512;       // 16.7M f32
  float* A    = (float*)alloc(ACTN * 4);
  float* Bb   = (float*)alloc(ACTN * 4);
  float* CBUF = (float*)alloc(ACTN * 4);
  float* EMB  = (float*)alloc((size_t)NBE * NIF * 128 * 4);
  float* LEAK = (float*)alloc((size_t)NBE * 512 * 4);
  float* MEAN = (float*)alloc(128 * 4);
  float* RSTD = (float*)alloc(128 * 4);
  _Float16* hLin = (_Float16*)alloc(1024 * 128 * 2);
  _Float16* hCW[6]; for (int i = 0; i < 6; ++i) hCW[i] = (_Float16*)alloc(128 * 128 * 4 * 2);
  _Float16* hOW = (_Float16*)alloc(128 * 128 * 3 * 2);
  _Float16* hWW = (_Float16*)alloc(128 * 128 * 2);
  _Float16* hBW = (_Float16*)alloc(128 * 128 * 2);
  _Float16* hEMBW[4]; for (int i = 0; i < 4; ++i) hEMBW[i] = (_Float16*)alloc(128 * 128 * 2);
  // scan stacks contiguous: [ol.ws0,ol.ws1,ol.ws2,ol.out_w,out.ws0..2,out.out_w]
  _Float16* hSCAN = (_Float16*)alloc(8 * 128 * 128 * 2);

  auto cvt = [&](const float* s, _Float16* d, int nel) {
    k_cvt<<<(nel + 255) / 256, 256, 0, stream>>>(s, d, nel);
  };
  cvt(up_linw, hLin, 1024 * 128);
  for (int i = 0; i < 6; ++i) cvt(up_cw[i], hCW[i], 128 * 128 * 4);
  cvt(up_ow, hOW, 128 * 128 * 3);
  cvt(p_w_w, hWW, 128 * 128);
  cvt(p_b_w, hBW, 128 * 128);
  for (int i = 0; i < 3; ++i) {
    cvt(e_ws[i], hEMBW[i], 16384);
    cvt(o_ws[i], hSCAN + (size_t)i * 16384, 16384);
    cvt(u_ws[i], hSCAN + (size_t)(4 + i) * 16384, 16384);
  }
  cvt(e_outw, hEMBW[3], 16384);
  cvt(o_outw, hSCAN + (size_t)3 * 16384, 16384);
  cvt(u_outw, hSCAN + (size_t)7 * 16384, 16384);

  // up-linear: (256,128) @ (1024,128)^T  -> A viewed as (256,128ch,8)
  k_gemm_bias<<<dim3(NBE / 16, 8), 256, 0, stream>>>(embedding, hLin, up_linb, A, 1024);

  // 6 upsampling conv layers with cross-batch BN + leaky
  float* cur = A; float* nxt = Bb;
  int L = 8;
  for (int i = 0; i < 6; ++i) {
    k_conv_dil<<<dim3((NBE * L) / 16, 2), 256, 0, stream>>>(cur, hCW[i], up_cb[i], nxt, L);
    L *= 2;
    k_bn_stats<<<128, 256, 0, stream>>>(nxt, L, MEAN, RSTD);
    k_bn_apply<<<(unsigned)((size_t)NBE * 128 * L / 256), 256, 0, stream>>>(nxt, L, MEAN, RSTD, up_bns[i], up_bnb[i]);
    float* t2 = cur; cur = nxt; nxt = t2;
  }
  // final conv: cur(=A) -> Bb holds d (256,128,512)
  k_conv_final<<<(NBE * 512) / 16, 256, 0, stream>>>(cur, hOW, up_ob, Bb);

  // unit-norm over channels + transpose + fused leak
  k_unorm_leak<<<NBE * 512, 128, 0, stream>>>(Bb, p_l_w, p_l_b, CBUF, LEAK);

  // per-frame modulation: weights -> A, biases -> Bb
  k_gemm_bias<<<dim3((NBE * 512) / 16, 1), 256, 0, stream>>>(CBUF, hWW, p_w_b, A, 128);
  k_gemm_bias<<<dim3((NBE * 512) / 16, 1), 256, 0, stream>>>(CBUF, hBW, p_b_b, Bb, 128);

  // impulse frames -> embed stack
  k_frames<<<(NBE * NIF) / 16, 256, 0, stream>>>(impulse,
      hEMBW[0], hEMBW[1], hEMBW[2], hEMBW[3],
      e_bs[0], e_bs[1], e_bs[2], e_lns[0], e_lns[1], e_lns[2],
      e_lnb[0], e_lnb[1], e_lnb[2], e_outb, EMB);

  // recurrent scan with LDS-resident stacks + fused overlap-add
  hipFuncSetAttribute(reinterpret_cast<const void*>(k_scan),
                      hipFuncAttributeMaxDynamicSharedMemorySize, SCAN_SMEM_BYTES);
  k_scan<<<NBE / 16, 256, SCAN_SMEM_BYTES, stream>>>(EMB, A, Bb, LEAK, hSCAN,
      o_bs[0], o_bs[1], o_bs[2], o_lns[0], o_lns[1], o_lns[2],
      o_lnb[0], o_lnb[1], o_lnb[2], o_outb,
      u_bs[0], u_bs[1], u_bs[2], u_lns[0], u_lns[1], u_lns[2],
      u_lnb[0], u_lnb[1], u_lnb[2], u_outb,
      (float*)d_out);
}
